// row_attention_maxindex_91139206021610
// MI455X (gfx1250) — matrix operand order repaired and checked by reading, not compiled, not run on hardware
//
#include <hip/hip_runtime.h>
#include <hip/hip_bf16.h>

// ---------------------------------------------------------------------------
// row_attention_maxindex for MI455X (gfx1250, wave32, WMMA f16 16x16x32)
// Flash-attention formulation: never materialize the 4096x4096 attention
// matrix; soft-argmax column expectation is an extra "virtual channel" of g.
// v2: 4 waves / workgroup share key tiles staged in LDS with double-buffered
//     GLOBAL_LOAD_ASYNC_TO_LDS (ASYNCcnt) -> 4x less L2 traffic.
// ---------------------------------------------------------------------------

typedef _Float16 h8  __attribute__((ext_vector_type(8)));
typedef _Float16 h16 __attribute__((ext_vector_type(16)));
typedef float    f8  __attribute__((ext_vector_type(8)));
typedef float    f4  __attribute__((ext_vector_type(4)));

#define B_   4
#define C_   256
#define REP  128
#define H_   32
#define W_   128
#define HW   4096      // H*W
#define GC   144       // g channels: 128 value + 1 col-index + 15 zero pad
#define EPSV 1e-5f

// ---- gfx1250 async global->LDS staging (guarded, with synchronous fallback)
#if defined(__has_builtin)
#if __has_builtin(__builtin_amdgcn_global_load_async_to_lds_b128) && \
    __has_builtin(__builtin_amdgcn_global_load_async_to_lds_b64)
#define HAVE_ASYNC 1
#endif
#endif
#ifndef HAVE_ASYNC
#define HAVE_ASYNC 0
#endif

#if HAVE_ASYNC
// exact param types per hipcc diagnostic: gcc-vector int pointers,
// global (AS1) source, LDS (AS3) destination, imm offset, imm cpol
typedef int av4i __attribute__((vector_size(16)));
typedef int av2i __attribute__((vector_size(8)));
#if __has_builtin(__builtin_amdgcn_s_wait_asynccnt)
#define WAIT_ASYNC(n) __builtin_amdgcn_s_wait_asynccnt(n)
#else
#define WAIT_ASYNC(n) asm volatile("s_wait_asynccnt %0" ::"i"(n) : "memory")
#endif
#else
#define WAIT_ASYNC(n) ((void)0)
#endif

static __device__ __forceinline__ void stage16B(const _Float16* g, _Float16* l) {
#if HAVE_ASYNC
  __builtin_amdgcn_global_load_async_to_lds_b128(
      (__attribute__((address_space(1))) av4i*)(g),
      (__attribute__((address_space(3))) av4i*)(l), 0, 0);
#else
  *(f4*)l = *(const f4*)g;
#endif
}
static __device__ __forceinline__ void stage8B(const _Float16* g, _Float16* l) {
#if HAVE_ASYNC
  __builtin_amdgcn_global_load_async_to_lds_b64(
      (__attribute__((address_space(1))) av2i*)(g),
      (__attribute__((address_space(3))) av2i*)(l), 0, 0);
#else
  *(unsigned long long*)l = *(const unsigned long long*)g;
#endif
}

static __device__ __forceinline__ h16 cat8(h8 lo, h8 hi) {
  return __builtin_shufflevector(lo, hi, 0,1,2,3,4,5,6,7,8,9,10,11,12,13,14,15);
}

static __device__ __forceinline__ f8 wmma16(h16 a, h16 b, f8 c) {
  // D = A(16x32 f16) * B(32x16 f16) + C(16x16 f32)
  return __builtin_amdgcn_wmma_f32_16x16x32_f16(
      /*neg_a=*/false, a, /*neg_b=*/false, b,
      /*c_mod=*/(short)0, c, /*reuse_a=*/false, /*reuse_b=*/false);
}

// ---------------------------------------------------------------------------
// 1) Projections: theta/phi/g 1x1 convs -> f16 tensors laid out for WMMA
//    thetaH[s][b][q][r]   (A-fragment friendly: rows of 128 contiguous halfs)
//    phiH  [s][b][key][r] (B-frag for S: lane = fixed key, 16 contiguous reps)
//    gT    [s][b][ch][key](B-frag for PV: lane = fixed ch, 16 contiguous keys)
//    gT channel 128 holds col_idx(key)=key%W; channels 129..143 are zero.
// ---------------------------------------------------------------------------
__global__ void proj_kernel(const float* __restrict__ left,
                            const float* __restrict__ right,
                            const float* __restrict__ pre_l,
                            const float* __restrict__ pre_r,
                            const float* __restrict__ query_l,
                            const float* __restrict__ key_l,
                            const float* __restrict__ query_r,
                            const float* __restrict__ key_r,
                            const float* __restrict__ theta_w,
                            const float* __restrict__ theta_b,
                            const float* __restrict__ phi_w,
                            const float* __restrict__ phi_b,
                            const float* __restrict__ g_w,
                            const float* __restrict__ g_b,
                            _Float16* __restrict__ thetaH,
                            _Float16* __restrict__ phiH,
                            _Float16* __restrict__ gT) {
  const int q = blockIdx.x;      // spatial position (query for theta, key for phi/g)
  const int b = blockIdx.y;
  const int s = blockIdx.z;      // side
  const int r = threadIdx.x;     // REP channel 0..127

  const float* xq  = s ? right   : left;
  const float* xkv = s ? left    : right;
  const float* pre = s ? pre_r   : pre_l;
  const float* qry = s ? query_r : query_l;
  const float* kft = s ? key_l   : key_r;   // key features come from the OTHER side

  float at = theta_b[r];
  float ap = phi_b[r];
  float ag = g_b[r];
  const size_t xoff = (size_t)b * C_ * HW + q;
  #pragma unroll 4
  for (int c = 0; c < C_; ++c) {
    float vq  = xq [xoff + (size_t)c * HW];
    float vkv = xkv[xoff + (size_t)c * HW];
    at = fmaf(vq,  theta_w[r * (C_ + 1) + c], at);
    ap = fmaf(vkv, phi_w  [r * C_ + c],       ap);
    ag = fmaf(vkv, g_w    [r * C_ + c],       ag);
  }
  at = fmaf(pre[(size_t)b * HW + q] * (1.0f / W_), theta_w[r * (C_ + 1) + C_], at);
  at += qry[((size_t)b * REP + r) * HW + q];
  ap += kft[((size_t)b * REP + r) * HW + q];

  const size_t sb = (size_t)s * B_ + b;
  thetaH[(sb * HW + q) * REP + r] = (_Float16)at;
  phiH  [(sb * HW + q) * REP + r] = (_Float16)ap;
  gT[(sb * GC + r) * HW + q] = (_Float16)ag;
  if (r < GC - REP) {
    gT[(sb * GC + REP + r) * HW + q] =
        (r == 0) ? (_Float16)(float)(q & (W_ - 1)) : (_Float16)0.0f;
  }
}

// ---------------------------------------------------------------------------
// 2) Flash attention: 4 waves per workgroup; each wave owns one 16-query tile,
//    all waves share double-buffered key tiles (phi 8KB + g_ext 9KB) staged
//    in LDS via async global->LDS loads. Per 32-key tile per wave:
//    8 WMMA for S = theta*phi, online softmax, 9 WMMA for acc += P * g_ext.
// ---------------------------------------------------------------------------
#define NSTAGE 17   // async ops per wave per key tile: 8 x B128 (phi) + 9 x B64 (g)

__global__ void __launch_bounds__(128)
attn_kernel(const _Float16* __restrict__ thetaH,
            const _Float16* __restrict__ phiH,
            const _Float16* __restrict__ gT,
            _Float16* __restrict__ afterH,     // [s][b][q][rep] f16
            float* __restrict__ out_index) {   // d_out + 2*B*C*HW
  __shared__ __align__(16) _Float16 phiLds[2][32][REP];   // [buf][key][rep] 16KB
  __shared__ __align__(16) _Float16 gLds[2][GC][32];      // [buf][ch][key]  18KB
  __shared__ __align__(16) _Float16 Plds[4][16][40];      // per-wave P transpose

  const int b  = blockIdx.y;
  const int s  = blockIdx.z;
  const int t  = threadIdx.x;       // 0..127
  const int w  = t >> 5;            // wave 0..3
  const int L  = t & 31;            // lane
  const int lh = L >> 4;            // lane half
  const int ll = L & 15;

  const size_t sb = (size_t)s * B_ + b;
  const _Float16* thetaP = thetaH + sb * (size_t)HW * REP;
  const _Float16* phiP   = phiH   + sb * (size_t)HW * REP;
  const _Float16* gP     = gT     + sb * (size_t)GC * HW;
  const int q0 = (blockIdx.x * 4 + w) * 16;   // this wave's query tile

  // A-fragments of the 16x128 theta tile (kept live across the whole K loop).
  // A k order used here (B matches): lane ll = row M; elem e -> K = lh*16 + e
  h16 aF[4];
  {
    const _Float16* rp = thetaP + (size_t)(q0 + ll) * REP;
    #pragma unroll
    for (int kc = 0; kc < 4; ++kc) {
      const _Float16* p = rp + kc * 32 + lh * 16;
      aF[kc] = cat8(*(const h8*)(p), *(const h8*)(p + 8));
    }
  }

  const f8 zero8 = {};
  f8 acc[9];
  #pragma unroll
  for (int tt = 0; tt < 9; ++tt) acc[tt] = zero8;
  float m[8], l[8];
  #pragma unroll
  for (int r = 0; r < 8; ++r) { m[r] = -1e30f; l[r] = 0.0f; }

  // ---- stage key tile (32 keys) into LDS buffer `buf` (cooperative, async)
  auto stage_tile = [&](int buf, int n0) {
    // phi tile: 32 rows x 128 reps = 8192B; loop copies 16384B contiguous (8 x B128 per thread)
    const _Float16* pg = phiP + (size_t)n0 * REP;
    _Float16* pl = &phiLds[buf][0][0];
    #pragma unroll
    for (int k = 0; k < 8; ++k) {
      int c = t + k * 128;                   // 16B chunk id, 0..1023
      stage16B(pg + c * 8, pl + c * 8);      // 8 halfs = 16B
    }
    // g tile: 144 rows x 32 keys (64B rows, strided in global) -> 9 x B64
    #pragma unroll
    for (int k = 0; k < 9; ++k) {
      int c = t + k * 128;                   // 8B chunk id, 0..1151
      int ch = c >> 3, part = c & 7;         // 8 chunks per 64B row
      stage8B(gP + (size_t)ch * HW + n0 + part * 4,
              &gLds[buf][ch][part * 4]);
    }
  };

  stage_tile(0, 0);

  const int NT = HW / 32;    // 128 key tiles
  for (int it = 0; it < NT; ++it) {
    const int cur = it & 1;
    if (it + 1 < NT) {
      stage_tile(cur ^ 1, (it + 1) * 32);    // prefetch next tile
      WAIT_ASYNC(NSTAGE);                    // current tile's loads complete
    } else {
      WAIT_ASYNC(0);
    }
    __syncthreads();                         // tile `it` visible to all waves

    // ---- S = theta * phi for 32 keys (two 16x16 D tiles) ----
    f8 s0 = zero8, s1 = zero8;
    #pragma unroll
    for (int kc = 0; kc < 4; ++kc) {
      // B k order used here (matches A): lane ll = col N (key); K = lh*16 + e contiguous
      const _Float16* p0 = &phiLds[cur][ll][kc * 32 + lh * 16];
      const _Float16* p1 = &phiLds[cur][16 + ll][kc * 32 + lh * 16];
      s0 = wmma16(aF[kc], *(const h16*)p0, s0);
      s1 = wmma16(aF[kc], *(const h16*)p1, s1);
    }

    // ---- online softmax (row M = lh*8 + r held across the 16 lanes of a half)
    float scale[8], p0v[8], p1v[8];
    #pragma unroll
    for (int r = 0; r < 8; ++r) {
      float mx = fmaxf(s0[r], s1[r]);
      mx = fmaxf(mx, __shfl_xor(mx, 1));
      mx = fmaxf(mx, __shfl_xor(mx, 2));
      mx = fmaxf(mx, __shfl_xor(mx, 4));
      mx = fmaxf(mx, __shfl_xor(mx, 8));
      float mn = fmaxf(m[r], mx);
      scale[r] = __expf(m[r] - mn);
      p0v[r] = __expf(s0[r] - mn);
      p1v[r] = __expf(s1[r] - mn);
      float rs = p0v[r] + p1v[r];
      rs += __shfl_xor(rs, 1);
      rs += __shfl_xor(rs, 2);
      rs += __shfl_xor(rs, 4);
      rs += __shfl_xor(rs, 8);
      l[r] = l[r] * scale[r] + rs;
      m[r] = mn;
    }
    #pragma unroll
    for (int tt = 0; tt < 9; ++tt) {
      #pragma unroll
      for (int r = 0; r < 8; ++r) acc[tt][r] *= scale[r];
    }

    // ---- transpose P (D layout) into an A fragment via per-wave LDS patch
    //      (within-wave: LDS ops are in-order, no barrier needed)
    #pragma unroll
    for (int r = 0; r < 8; ++r) {
      Plds[w][lh * 8 + r][ll]      = (_Float16)p0v[r];
      Plds[w][lh * 8 + r][16 + ll] = (_Float16)p1v[r];
    }
    h16 pF = cat8(*(const h8*)(&Plds[w][ll][lh * 16]),
                  *(const h8*)(&Plds[w][ll][lh * 16 + 8]));

    // ---- acc += P(16x32) * g_ext(32x144): 9 channel tiles ----
    #pragma unroll
    for (int nt = 0; nt < 9; ++nt) {
      const _Float16* gp = &gLds[cur][nt * 16 + ll][lh * 16];
      acc[nt] = wmma16(pF, *(const h16*)gp, acc[nt]);
    }

    __syncthreads();   // all waves done reading buf `cur` before it is refilled
  }

  // ---- epilogue: normalize, emit after (f16) and soft-argmax index ----
  float inv[8];
  #pragma unroll
  for (int r = 0; r < 8; ++r) inv[r] = 1.0f / l[r];
  #pragma unroll
  for (int nt = 0; nt < 8; ++nt) {
    #pragma unroll
    for (int r = 0; r < 8; ++r) {
      int q = q0 + lh * 8 + r;               // D row M = r + lh*8
      afterH[(sb * HW + q) * REP + nt * 16 + ll] =
          (_Float16)(acc[nt][r] * inv[r]);
    }
  }
  if (ll == 0) {                              // channel tile 8, col 0 = col-idx
    #pragma unroll
    for (int r = 0; r < 8; ++r) {
      int q = q0 + lh * 8 + r;
      float expcol = acc[8][r] * inv[r];
      out_index[sb * HW + q] = (float)(q & (W_ - 1)) - expcol;
    }
  }
}

// ---------------------------------------------------------------------------
// 3) up 1x1 conv via WMMA: y[s][b][co][q] = after[q,:] . up_w[co,:] + up_b[co]
//    up_w's native [co][rep] layout IS the B-fragment layout.
// ---------------------------------------------------------------------------
__global__ void __launch_bounds__(32)
upconv_kernel(const _Float16* __restrict__ afterH,
              const _Float16* __restrict__ upWh,    // [C][REP] f16
              const float* __restrict__ up_b,
              float* __restrict__ y) {
  const int qt = blockIdx.x, b = blockIdx.y, s = blockIdx.z;
  const int L = threadIdx.x, lh = L >> 4, ll = L & 15;
  const size_t sb = (size_t)s * B_ + b;
  const int q0 = qt * 16;

  h16 aF[4];
  const _Float16* rp = afterH + (sb * HW + q0 + ll) * (size_t)REP;
  #pragma unroll
  for (int kc = 0; kc < 4; ++kc) {
    const _Float16* p = rp + kc * 32 + lh * 16;
    aF[kc] = cat8(*(const h8*)p, *(const h8*)(p + 8));
  }

  const f8 zero8 = {};
  #pragma unroll
  for (int nt = 0; nt < C_ / 16; ++nt) {
    f8 d = zero8;
    #pragma unroll
    for (int kc = 0; kc < 4; ++kc) {
      const _Float16* wp = upWh + (size_t)(nt * 16 + ll) * REP + kc * 32 + lh * 16;
      d = wmma16(aF[kc], *(const h16*)wp, d);
    }
    int co = nt * 16 + ll;
    float bias = up_b[co];
    #pragma unroll
    for (int r = 0; r < 8; ++r) {
      int q = q0 + lh * 8 + r;
      y[(sb * C_ + co) * (size_t)HW + q] = d[r] + bias;
    }
  }
}

// ---------------------------------------------------------------------------
// 4) BatchNorm train-mode statistics per (side, channel) over B*HW samples
// ---------------------------------------------------------------------------
__global__ void bnstats_kernel(const float* __restrict__ y,
                               float* __restrict__ stats) {  // [s][C][2]
  const int co = blockIdx.x, s = blockIdx.y, tid = threadIdx.x;
  __shared__ float ssum[256], ssq[256];
  float sum = 0.0f, sq = 0.0f;
  for (int i = tid; i < B_ * HW; i += 256) {
    int b = i >> 12;          // / HW
    int q = i & (HW - 1);
    float v = y[(((size_t)s * B_ + b) * C_ + co) * HW + q];
    sum += v;
    sq  += v * v;
  }
  ssum[tid] = sum; ssq[tid] = sq;
  __syncthreads();
  for (int off = 128; off > 0; off >>= 1) {
    if (tid < off) { ssum[tid] += ssum[tid + off]; ssq[tid] += ssq[tid + off]; }
    __syncthreads();
  }
  if (tid == 0) {
    const float n = (float)(B_ * HW);
    float mean = ssum[0] / n;
    float var  = ssq[0] / n - mean * mean;
    stats[(s * C_ + co) * 2 + 0] = mean;
    stats[(s * C_ + co) * 2 + 1] = rsqrtf(var + EPSV);
  }
}

// ---------------------------------------------------------------------------
// 5) Finalize: out = x_q + gamma*(y - mean)*rstd + beta
// ---------------------------------------------------------------------------
__global__ void finalize_kernel(const float* __restrict__ left,
                                const float* __restrict__ right,
                                const float* __restrict__ y,
                                const float* __restrict__ stats,
                                const float* __restrict__ gamma,
                                const float* __restrict__ beta,
                                float* __restrict__ out) {
  const int s = blockIdx.y;
  const size_t idx = (size_t)blockIdx.x * 256 + threadIdx.x;  // < B*C*HW
  const int co = (int)((idx >> 12) & (C_ - 1));               // HW=4096, C=256
  const float* xq = s ? right : left;
  float v    = y[(size_t)s * B_ * C_ * HW + idx];
  float mean = stats[(s * C_ + co) * 2 + 0];
  float rstd = stats[(s * C_ + co) * 2 + 1];
  out[(size_t)s * B_ * C_ * HW + idx] =
      xq[idx] + gamma[co] * (v - mean) * rstd + beta[co];
}

// weight f32 -> f16 conversion for the up-projection B matrix
__global__ void cvtw_kernel(const float* __restrict__ w,
                            _Float16* __restrict__ wh, int n) {
  int i = blockIdx.x * 256 + threadIdx.x;
  if (i < n) wh[i] = (_Float16)w[i];
}

// ---------------------------------------------------------------------------
extern "C" void kernel_launch(void* const* d_in, const int* in_sizes, int n_in,
                              void* d_out, int out_size, void* d_ws, size_t ws_size,
                              hipStream_t stream) {
  const float* left    = (const float*)d_in[0];
  const float* right   = (const float*)d_in[1];
  const float* pre_l   = (const float*)d_in[2];
  const float* pre_r   = (const float*)d_in[3];
  const float* query_l = (const float*)d_in[4];
  const float* key_l   = (const float*)d_in[5];
  const float* query_r = (const float*)d_in[6];
  const float* key_r   = (const float*)d_in[7];
  const float* theta_w = (const float*)d_in[8];
  const float* theta_b = (const float*)d_in[9];
  const float* phi_w   = (const float*)d_in[10];
  const float* phi_b   = (const float*)d_in[11];
  const float* g_w     = (const float*)d_in[12];
  const float* g_b     = (const float*)d_in[13];
  const float* up_w    = (const float*)d_in[14];
  const float* up_b    = (const float*)d_in[15];
  const float* bn_g    = (const float*)d_in[16];
  const float* bn_b    = (const float*)d_in[17];

  float* out = (float*)d_out;
  char* ws = (char*)d_ws;

  // workspace layout (all offsets 32B aligned)
  const size_t szThetaH = 2ull * B_ * HW * REP * sizeof(_Float16); // 8 MB
  const size_t szPhiH   = szThetaH;                                // 8 MB
  const size_t szGT     = 2ull * B_ * GC * HW * sizeof(_Float16);  // 9 MB
  const size_t szAfter  = szThetaH;                                // 8 MB
  const size_t szUpWh   = (size_t)C_ * REP * sizeof(_Float16);     // 64 KB
  const size_t szY      = 2ull * B_ * C_ * HW * sizeof(float);     // 32 MB

  size_t o = 0;
  _Float16* thetaH = (_Float16*)(ws + o); o += szThetaH;
  _Float16* phiH   = (_Float16*)(ws + o); o += szPhiH;
  _Float16* gT     = (_Float16*)(ws + o); o += szGT;
  _Float16* afterH = (_Float16*)(ws + o); o += szAfter;
  _Float16* upWh   = (_Float16*)(ws + o); o += szUpWh;
  float*    yBuf   = (float*)   (ws + o); o += szY;
  float*    stats  = (float*)   (ws + o); o += 2ull * C_ * 2 * sizeof(float);
  (void)o; (void)ws_size; (void)in_sizes; (void)n_in; (void)out_size;

  float* out_index = out + 2ull * B_ * C_ * HW;  // after l_n, r_n

  // 0) convert up_w to f16 (B-fragment-native [co][rep] layout)
  cvtw_kernel<<<dim3((C_ * REP + 255) / 256), dim3(256), 0, stream>>>(
      up_w, upWh, C_ * REP);

  // 1) projections for both sides
  proj_kernel<<<dim3(HW, B_, 2), dim3(REP), 0, stream>>>(
      left, right, pre_l, pre_r, query_l, key_l, query_r, key_r,
      theta_w, theta_b, phi_w, phi_b, g_w, g_b, thetaH, phiH, gT);

  // 2) flash attention + soft-argmax (WMMA + async LDS staging)
  attn_kernel<<<dim3(HW / 64, B_, 2), dim3(128), 0, stream>>>(
      thetaH, phiH, gT, afterH, out_index);

  // 3) up-projection via WMMA
  upconv_kernel<<<dim3(HW / 16, B_, 2), dim3(32), 0, stream>>>(
      afterH, upWh, up_b, yBuf);

  // 4) batchnorm statistics
  bnstats_kernel<<<dim3(C_, 2), dim3(256), 0, stream>>>(yBuf, stats);

  // 5) normalize + residual
  finalize_kernel<<<dim3((B_ * C_ * HW) / 256, 2), dim3(256), 0, stream>>>(
      left, right, yBuf, stats, bn_g, bn_b, out);
}
